// QuantizedCpuLinear_32418413150232
// MI455X (gfx1250) — matrix operand order repaired and checked by reading, not compiled, not run on hardware
//
#include <hip/hip_runtime.h>

typedef __attribute__((ext_vector_type(16))) _Float16 v16h;
typedef __attribute__((ext_vector_type(8)))  _Float16 v8h;
typedef __attribute__((ext_vector_type(4)))  _Float16 v4h;
typedef __attribute__((ext_vector_type(2)))  _Float16 v2h;
typedef __attribute__((ext_vector_type(8)))  float    v8f;
typedef __attribute__((ext_vector_type(4)))  float    v4f;
typedef __attribute__((ext_vector_type(2)))  unsigned int u32x2;
typedef __attribute__((ext_vector_type(4)))  unsigned int u32x4;

namespace {
constexpr int BM   = 128;          // block tile M
constexpr int BN   = 256;          // block tile N
constexpr int BK   = 32;           // k-tile (= one WMMA k-step)
constexpr int PAD  = 8;            // halves of padding per LDS row (bank stagger)
constexpr int LDT  = BK + PAD;     // 40 halves = 80 B row stride
constexpr int MDIM = 8192;         // B*S
constexpr int NDIM = 4096;         // out features
constexpr int KDIM = 4096;         // in features
constexpr int NKT  = KDIM / BK;    // 128 k-iterations
}

// int8x4 (packed in a u32) -> 4 exact f16, via exponent trick:
//   b^0x80 maps b to b+128 in [0,255]; 0x6400|u encodes 1024+u in f16;
//   subtract 1152 -> b.  5 VALU per 4 elements (xor + 2 perm + 2 pk_add).
__device__ inline v4h int8x4_to_half4(unsigned int w) {
    unsigned int u = w ^ 0x80808080u;
    unsigned int h01 = __builtin_amdgcn_perm(0x64646464u, u, 0x05010400u); // {b0,64,b1,64}
    unsigned int h23 = __builtin_amdgcn_perm(0x64646464u, u, 0x05030402u); // {b2,64,b3,64}
    v2h a = __builtin_bit_cast(v2h, h01);
    v2h b = __builtin_bit_cast(v2h, h23);
    const v2h off = {(_Float16)1152.0f, (_Float16)1152.0f};
    a = a - off;
    b = b - off;
    v4h r;
    r[0] = a[0]; r[1] = a[1]; r[2] = b[0]; r[3] = b[1];
    return r;
}

// ---------------- pre-pass: Wq int8 -> f16 (exact), 16 elems/thread ----------------
__global__ __launch_bounds__(256)
void convert_w_f16(const unsigned int* __restrict__ Wq, _Float16* __restrict__ Wf) {
    const long idx = (long)blockIdx.x * 256 + threadIdx.x;   // one 16-byte group
    u32x4 w = ((const u32x4*)Wq)[idx];
    v4h h0 = int8x4_to_half4(w.x);
    v4h h1 = int8x4_to_half4(w.y);
    v4h h2 = int8x4_to_half4(w.z);
    v4h h3 = int8x4_to_half4(w.w);
    _Float16* out = Wf + idx * 16;
    union { v8h v; v4h q[2]; } lo, hi;
    lo.q[0] = h0; lo.q[1] = h1;
    hi.q[0] = h2; hi.q[1] = h3;
    *(v8h*)(out)     = lo.v;
    *(v8h*)(out + 8) = hi.v;
}

// ---------------- main GEMM ----------------
// PREW = true : W already f16 in Wsrc (d_ws), B staging is a pure copy.
// PREW = false: W is int8, dequant inline with the perm trick (fallback).
template <bool PREW>
__global__ __launch_bounds__(256)
void qlinear_wmma(const float* __restrict__ X,
                  const void* __restrict__ Wsrc,
                  const float* __restrict__ Sc,
                  float* __restrict__ Y)
{
    __shared__ _Float16 As[2][BM][LDT];   // x tile, f16, [M][K]   (20.0 KB)
    __shared__ _Float16 Bs[2][BN][LDT];   // w tile, f16, [N][K]   (40.0 KB)

    const int tid   = threadIdx.x;
    const int lane  = tid & 31;
    const int wv    = tid >> 5;
    const int waveM = wv >> 2;            // 0..1  : 64-row slab
    const int waveN = wv & 3;             // 0..3  : 64-col slab
    const int half  = lane >> 4;          // 0..1
    const int l15   = lane & 15;

    const int bm = blockIdx.y * BM;
    const int bn = blockIdx.x * BN;

    // ---- cooperative staging indices ----
    const int arow = tid >> 3;            // 8 float4 per 32-wide row (A)
    const int acol = (tid & 7) * 4;
    // B: one 32-element row per thread (64 B along K)

    const float* aptr = X + (long)(bm + arow) * KDIM + acol;
    const _Float16*    wfp = (const _Float16*)Wsrc    + (long)(bn + tid) * KDIM;
    const signed char* wqp = (const signed char*)Wsrc + (long)(bn + tid) * KDIM;

    v4f   areg[4];
    v8h   bregh[4];   // PREW path: 4 x 16B along one row
    u32x4 bregq[2];   // int8 path: 32 bytes of one row

    auto gload = [&](int kt) {
#pragma unroll
        for (int i = 0; i < 4; ++i)
            areg[i] = *(const v4f*)(aptr + kt * BK + (long)i * 32 * KDIM);
        if (PREW) {
#pragma unroll
            for (int g = 0; g < 4; ++g)
                bregh[g] = *(const v8h*)(wfp + kt * BK + g * 8);
        } else {
#pragma unroll
            for (int g = 0; g < 2; ++g)
                bregq[g] = *(const u32x4*)(wqp + kt * BK + g * 16);
        }
    };

    auto stage = [&](int buf) {
#pragma unroll
        for (int i = 0; i < 4; ++i) {
            v4h h;
#pragma unroll
            for (int j = 0; j < 4; ++j) h[j] = (_Float16)areg[i][j];
            *(v4h*)&As[buf][arow + i * 32][acol] = h;
        }
        if (PREW) {
#pragma unroll
            for (int g = 0; g < 4; ++g)
                *(v8h*)&Bs[buf][tid][g * 8] = bregh[g];
        } else {
#pragma unroll
            for (int g = 0; g < 2; ++g) {
                union { v8h v; v4h q[2]; } h0, h1;
                h0.q[0] = int8x4_to_half4(bregq[g].x);
                h0.q[1] = int8x4_to_half4(bregq[g].y);
                h1.q[0] = int8x4_to_half4(bregq[g].z);
                h1.q[1] = int8x4_to_half4(bregq[g].w);
                *(v8h*)&Bs[buf][tid][g * 16]     = h0.v;
                *(v8h*)&Bs[buf][tid][g * 16 + 8] = h1.v;
            }
        }
    };

    v8f acc[4][4];
#pragma unroll
    for (int i = 0; i < 4; ++i)
#pragma unroll
        for (int j = 0; j < 4; ++j)
#pragma unroll
            for (int e = 0; e < 8; ++e)
                acc[i][j][e] = 0.0f;

    union V16 { v16h v; v8h h[2]; };

    auto compute = [&](int buf) {
        const int r0 = waveM * 64;
        const int n0 = waveN * 64;
        // B fragments: lane holds column n, 16 contiguous K at 16*(lane>>4)
        V16 bfrag[4];
#pragma unroll
        for (int j = 0; j < 4; ++j) {
            const _Float16* p = &Bs[buf][n0 + j * 16 + l15][half * 16];
            bfrag[j].h[0] = *(const v8h*)(p);
            bfrag[j].h[1] = *(const v8h*)(p + 8);
        }
#pragma unroll
        for (int i = 0; i < 4; ++i) {
            // A fragment: lane holds row m, 16 contiguous K at 16*(lane>>4)
            V16 afrag;
            const _Float16* p = &As[buf][r0 + i * 16 + l15][half * 16];
            afrag.h[0] = *(const v8h*)(p);
            afrag.h[1] = *(const v8h*)(p + 8);
#pragma unroll
            for (int j = 0; j < 4; ++j)
                acc[i][j] = __builtin_amdgcn_wmma_f32_16x16x32_f16(
                    false, afrag.v, false, bfrag[j].v,
                    (short)0, acc[i][j], false, false);
        }
    };

    // ---- software pipeline: prefetch / compute / stage, double-buffered LDS ----
    gload(0);
    stage(0);
    __syncthreads();

    for (int kt = 0; kt < NKT; ++kt) {
        const int buf = kt & 1;
        if (kt + 1 < NKT) gload(kt + 1);   // issue next global loads
        compute(buf);                      // 16 WMMAs/wave hide the load latency
        if (kt + 1 < NKT) stage(buf ^ 1);  // stage into the other buffer
        __syncthreads();
    }

    // ---- epilogue: apply per-tensor scale in fp32, store ----
    const float s = Sc[0];
#pragma unroll
    for (int i = 0; i < 4; ++i) {
#pragma unroll
        for (int j = 0; j < 4; ++j) {
            const int m = bm + waveM * 64 + i * 16 + half * 8;
            const int n = bn + waveN * 64 + j * 16 + l15;
            float* out = Y + (long)m * NDIM + n;
#pragma unroll
            for (int r = 0; r < 8; ++r)
                out[(long)r * NDIM] = acc[i][j][r] * s;
        }
    }
}

extern "C" void kernel_launch(void* const* d_in, const int* in_sizes, int n_in,
                              void* d_out, int out_size, void* d_ws, size_t ws_size,
                              hipStream_t stream) {
    (void)in_sizes; (void)n_in; (void)out_size;
    const float*       x  = (const float*)d_in[0];
    const signed char* wq = (const signed char*)d_in[1];
    const float*       sc = (const float*)d_in[2];
    float*             y  = (float*)d_out;

    dim3 grid(NDIM / BN, MDIM / BM);   // (16, 64)
    dim3 block(256);                   // 8 wave32 waves

    constexpr size_t WBYTES = (size_t)NDIM * KDIM * sizeof(_Float16); // 32 MB

    if (ws_size >= WBYTES) {
        // Pre-pass: exact int8 -> f16 weight dequant into workspace (once per call).
        const int total_groups = NDIM * KDIM / 16;       // 16 elems per thread
        convert_w_f16<<<total_groups / 256, 256, 0, stream>>>(
            (const unsigned int*)wq, (_Float16*)d_ws);
        qlinear_wmma<true><<<grid, block, 0, stream>>>(x, d_ws, sc, y);
    } else {
        // Fallback: inline dequant with the perm trick.
        qlinear_wmma<false><<<grid, block, 0, stream>>>(x, wq, sc, y);
    }
}
